// GraphSAGE_60782377173253
// MI455X (gfx1250) — matrix operand order repaired and checked by reading, not compiled, not run on hardware
//
#include <hip/hip_runtime.h>

#define NNODES 50000
#define DIM    256
#define SNEI   25

typedef __bf16 bf16_t;
typedef __bf16 bf16x16 __attribute__((ext_vector_type(16)));
typedef float  f32x8   __attribute__((ext_vector_type(8)));
typedef unsigned int u32x4 __attribute__((ext_vector_type(4)));

struct U32x8 { u32x4 lo, hi; };

// ---- WMMA fragment loaders (CDNA5 16-bit layouts, ISA 7.12.2) ----
// A (16x32, MxK): lane holds row M=lane%16; lanes 0-15 cover K={0..7,16..23},
// lanes 16-31 cover K={8..15,24..31}. Two aligned b128 loads.
__device__ __forceinline__ bf16x16 load_a_frag(const bf16_t* rowptr, int k0, int laneHi) {
    const int kb = k0 + (laneHi << 3);
    U32x8 t;
    t.lo = *(const u32x4*)(rowptr + kb);
    t.hi = *(const u32x4*)(rowptr + kb + 16);
    return __builtin_bit_cast(bf16x16, t);
}
// B (32x16, KxN): lane holds column N=lane%16; lanes 0-15 K={0..7,16..23},
// lanes 16-31 K={8..15,24..31}. Two aligned b128 loads from transposed weights.
__device__ __forceinline__ bf16x16 load_b_frag(const bf16_t* colptr, int k0, int laneHi) {
    const int kb = k0 + (laneHi << 3);
    U32x8 t;
    t.lo = *(const u32x4*)(colptr + kb);
    t.hi = *(const u32x4*)(colptr + kb + 16);
    return __builtin_bit_cast(bf16x16, t);
}

// ---- Pool GEMM: Y = relu(X @ W + b), X[N,256] bf16, Wt[256,256] bf16 (transposed), Y bf16 ----
__global__ __launch_bounds__(256) void gemm_pool_relu(const bf16_t* __restrict__ X,
                                                      const bf16_t* __restrict__ Wt,
                                                      const float*  __restrict__ bias,
                                                      bf16_t* __restrict__ Y) {
    const int lane   = threadIdx.x & 31;
    const int wave   = threadIdx.x >> 5;
    const int tile   = blockIdx.x * 8 + wave;
    if (tile >= (NNODES / 16) * (DIM / 16)) return;
    const int rowBase = (tile >> 4) << 4;   // 16 col-tiles per row-block
    const int colBase = (tile & 15) << 4;
    const int laneHi  = (lane >> 4) & 1;
    const int m       = lane & 15;

    const bf16_t* arow = X  + (size_t)(rowBase + m) * DIM;
    const bf16_t* bcol = Wt + (size_t)(colBase + m) * DIM;

    f32x8 c = {};
#pragma unroll
    for (int k0 = 0; k0 < DIM; k0 += 32) {
        bf16x16 a = load_a_frag(arow, k0, laneHi);
        bf16x16 b = load_b_frag(bcol, k0, laneHi);
        c = __builtin_amdgcn_wmma_f32_16x16x32_bf16(false, a, false, b, (short)0, c, false, false);
    }

    const int   col    = colBase + m;
    const float bv     = bias[col];
    const int   rowTop = rowBase + laneHi * 8;   // C: vgpr r -> M=r (lanes 0-15) / M=r+8 (16-31)
#pragma unroll
    for (int r = 0; r < 8; ++r) {
        float v = c[r] + bv;
        v = v > 0.f ? v : 0.f;
        Y[(size_t)(rowTop + r) * DIM + col] = (bf16_t)v;
    }
}

// ---- FC GEMM over implicit concat: OUT = [X | AGG] @ Wfc + b (K=512) ----
template <bool RELU_BF16_OUT>
__global__ __launch_bounds__(256) void gemm_fc(const bf16_t* __restrict__ X,
                                               const bf16_t* __restrict__ AGG,
                                               const bf16_t* __restrict__ Wt,   // [256][512] transposed
                                               const float*  __restrict__ bias,
                                               bf16_t* __restrict__ outB,
                                               float*  __restrict__ outF) {
    const int lane   = threadIdx.x & 31;
    const int wave   = threadIdx.x >> 5;
    const int tile   = blockIdx.x * 8 + wave;
    if (tile >= (NNODES / 16) * (DIM / 16)) return;
    const int rowBase = (tile >> 4) << 4;
    const int colBase = (tile & 15) << 4;
    const int laneHi  = (lane >> 4) & 1;
    const int m       = lane & 15;

    const size_t rOff = (size_t)(rowBase + m) * DIM;
    const bf16_t* bcol = Wt + (size_t)(colBase + m) * (2 * DIM);

    f32x8 c = {};
#pragma unroll
    for (int kk = 0; kk < 2 * DIM; kk += 32) {
        const bf16_t* asrc = (kk < DIM) ? (X + rOff) : (AGG + rOff);
        const int kloc = kk & (DIM - 1);
        bf16x16 a = load_a_frag(asrc, kloc, laneHi);
        bf16x16 b = load_b_frag(bcol, kk, laneHi);
        c = __builtin_amdgcn_wmma_f32_16x16x32_bf16(false, a, false, b, (short)0, c, false, false);
    }

    const int   col    = colBase + m;
    const float bv     = bias[col];
    const int   rowTop = rowBase + laneHi * 8;
#pragma unroll
    for (int r = 0; r < 8; ++r) {
        float v = c[r] + bv;
        if (RELU_BF16_OUT) {
            v = v > 0.f ? v : 0.f;
            outB[(size_t)(rowTop + r) * DIM + col] = (bf16_t)v;
        } else {
            outF[(size_t)(rowTop + r) * DIM + col] = v;
        }
    }
}

// ---- Neighbor max-pool: AGG[n,d] = max_s H[idx[n,s], d] ----
// One wave per node; each lane gathers 8 channels via a single b128 load per
// neighbor (32 lanes x 16B = whole 512B row). Max done in f32 via bit tricks
// (exact for bf16 operands), repacked without any cvt.
__global__ __launch_bounds__(256) void aggmax(const bf16_t* __restrict__ H,
                                              const int*    __restrict__ idx,
                                              bf16_t* __restrict__ AGG) {
    const int wave = threadIdx.x >> 5;
    const int lane = threadIdx.x & 31;
    const int n    = blockIdx.x * 8 + wave;
    if (n >= NNODES) return;
    const int* row = idx + (size_t)n * SNEI;
    const int chan = lane * 8;                 // 8 bf16 channels per lane

    float acc[8];
#pragma unroll
    for (int i = 0; i < 8; ++i) acc[i] = -3.402823466e38f;

#pragma unroll 5
    for (int s = 0; s < SNEI; ++s) {
        const int j = row[s];                  // wave-uniform
        const u32x4 d = *(const u32x4*)(H + (size_t)j * DIM + chan);
#pragma unroll
        for (int w = 0; w < 4; ++w) {
            const unsigned int u = d[w];
            const float lo = __uint_as_float(u << 16);
            const float hi = __uint_as_float(u & 0xffff0000u);
            acc[2 * w]     = fmaxf(acc[2 * w],     lo);
            acc[2 * w + 1] = fmaxf(acc[2 * w + 1], hi);
        }
    }

    u32x4 out;
#pragma unroll
    for (int w = 0; w < 4; ++w) {
        out[w] = (__float_as_uint(acc[2 * w]) >> 16) |
                 (__float_as_uint(acc[2 * w + 1]) & 0xffff0000u);
    }
    *(u32x4*)(AGG + (size_t)n * DIM + chan) = out;
}

// ---- conversions ----
__global__ __launch_bounds__(256) void f32_to_bf16(const float* __restrict__ in,
                                                   bf16_t* __restrict__ out, int n) {
    const int i = blockIdx.x * 256 + threadIdx.x;
    if (i < n) out[i] = (bf16_t)in[i];
}

// W is [K x Nc] row-major f32; Wt[n*K + k] = W[k*Nc + n] as bf16
__global__ __launch_bounds__(256) void transpose_to_bf16(const float* __restrict__ W,
                                                         bf16_t* __restrict__ Wt,
                                                         int K, int Nc) {
    const int i = blockIdx.x * 256 + threadIdx.x;
    if (i < K * Nc) {
        const int n = i / K;
        const int k = i - n * K;
        Wt[i] = (bf16_t)W[(size_t)k * Nc + n];
    }
}

extern "C" void kernel_launch(void* const* d_in, const int* in_sizes, int n_in,
                              void* d_out, int out_size, void* d_ws, size_t ws_size,
                              hipStream_t stream) {
    const float* features = (const float*)d_in[0];
    const int*   neigh    = (const int*)  d_in[1];
    const float* Wp0  = (const float*)d_in[2];
    const float* bp0  = (const float*)d_in[3];
    const float* Wfc0 = (const float*)d_in[4];
    const float* bfc0 = (const float*)d_in[5];
    const float* Wp1  = (const float*)d_in[6];
    const float* bp1  = (const float*)d_in[7];
    const float* Wfc1 = (const float*)d_in[8];
    const float* bfc1 = (const float*)d_in[9];
    float* out = (float*)d_out;

    const size_t ND = (size_t)NNODES * DIM;           // 12.8M elements
    char*  ws  = (char*)d_ws;
    size_t off = 0;
    auto carve = [&](size_t bytes) -> void* {
        void* p = ws + off;
        off = (off + bytes + 255) & ~(size_t)255;
        return p;
    };
    bf16_t* X0    = (bf16_t*)carve(ND * 2);
    bf16_t* X1    = (bf16_t*)carve(ND * 2);
    bf16_t* H     = (bf16_t*)carve(ND * 2);
    bf16_t* AGG   = (bf16_t*)carve(ND * 2);
    bf16_t* Wp0t  = (bf16_t*)carve((size_t)DIM * DIM * 2);
    bf16_t* Wfc0t = (bf16_t*)carve((size_t)DIM * 2 * DIM * 2);
    bf16_t* Wp1t  = (bf16_t*)carve((size_t)DIM * DIM * 2);
    bf16_t* Wfc1t = (bf16_t*)carve((size_t)DIM * 2 * DIM * 2);

    const int gemmBlocks = (NNODES / 16) * (DIM / 16) / 8;   // 6250
    const int aggBlocks  = NNODES / 8;                       // 6250

    // precision/layout prep
    f32_to_bf16<<<(int)((ND + 255) / 256), 256, 0, stream>>>(features, X0, (int)ND);
    transpose_to_bf16<<<(DIM * DIM + 255) / 256, 256, 0, stream>>>(Wp0, Wp0t, DIM, DIM);
    transpose_to_bf16<<<(2 * DIM * DIM + 255) / 256, 256, 0, stream>>>(Wfc0, Wfc0t, 2 * DIM, DIM);
    transpose_to_bf16<<<(DIM * DIM + 255) / 256, 256, 0, stream>>>(Wp1, Wp1t, DIM, DIM);
    transpose_to_bf16<<<(2 * DIM * DIM + 255) / 256, 256, 0, stream>>>(Wfc1, Wfc1t, 2 * DIM, DIM);

    // layer 0
    gemm_pool_relu<<<gemmBlocks, 256, 0, stream>>>(X0, Wp0t, bp0, H);
    aggmax<<<aggBlocks, 256, 0, stream>>>(H, neigh, AGG);
    gemm_fc<true><<<gemmBlocks, 256, 0, stream>>>(X0, AGG, Wfc0t, bfc0, X1, nullptr);

    // layer 1
    gemm_pool_relu<<<gemmBlocks, 256, 0, stream>>>(X1, Wp1t, bp1, H);
    aggmax<<<aggBlocks, 256, 0, stream>>>(H, neigh, AGG);
    gemm_fc<false><<<gemmBlocks, 256, 0, stream>>>(X1, AGG, Wfc1t, bfc1, nullptr, out);
}
